// PointNetReconstruct_31525059952828
// MI455X (gfx1250) — matrix operand order repaired and checked by reading, not compiled, not run on hardware
//
#include <hip/hip_runtime.h>
#include <hip/hip_bf16.h>

// ---------------------------------------------------------------------------
// PointNet feature-propagation + 3-layer (1x1 conv + BN + ReLU) MLP, gfx1250.
// GEMMs run on V_WMMA_F32_16X16X32_BF16 (bf16 in, f32 accumulate), with the
// activation (B-matrix) tile staged in LDS once per block and shared by all
// waves; each wave computes a 64x32 output tile (4 m-tiles x 2 p-tiles).
// Conv bias is dropped: a per-channel constant shift cancels exactly in BN.
// ---------------------------------------------------------------------------

#define B_    8
#define N_    4096
#define S_    1024
#define M_    2048
#define D1_   128
#define D2_   256
#define CIN_  388
#define KPAD0 416          // 13 * 32, zero padded K for layer 0
#define P_    (B_ * N_)    // 32768 points
#define C0_   512
#define C1_   512
#define C2_   256
#define BN_EPS_ 1e-5f

typedef __attribute__((ext_vector_type(16))) __bf16 v16bf;
typedef __attribute__((ext_vector_type(8)))  __bf16 v8bf;
typedef __attribute__((ext_vector_type(4)))  __bf16 v4bf;
typedef __attribute__((ext_vector_type(8)))  float  v8f;

// ---------------------------------------------------------------------------
// Kernel 1: per-point 3-NN over xyz2 (S=1024) + argmin over feature keypoints
// (M=2048). Reference sets staged in LDS (44 KB). Writes knn idx/weights and
// the last 4+pad channels of the X matrix (bf16, [P, KPAD0] row-major).
// ---------------------------------------------------------------------------
__global__ void knn_prep_kernel(const float* __restrict__ xyz1,
                                const float* __restrict__ xyz2,
                                const float* __restrict__ feature,
                                __bf16* __restrict__ Xbf,
                                int* __restrict__ idx3,
                                float* __restrict__ w3) {
  __shared__ float sx[S_], sy[S_], sz[S_];
  __shared__ float sf[M_ * 4];
  const int tid   = threadIdx.x;
  const int b     = blockIdx.x >> 4;
  const int chunk = blockIdx.x & 15;

  for (int i = tid; i < S_; i += 256) {
    sx[i] = xyz2[(b * 3 + 0) * S_ + i];
    sy[i] = xyz2[(b * 3 + 1) * S_ + i];
    sz[i] = xyz2[(b * 3 + 2) * S_ + i];
  }
  for (int i = tid; i < M_ * 4; i += 256) sf[i] = feature[(size_t)b * M_ * 4 + i];
  __syncthreads();

  const int n = chunk * 256 + tid;
  const int p = b * N_ + n;
  const float px = xyz1[(b * 3 + 0) * N_ + n];
  const float py = xyz1[(b * 3 + 1) * N_ + n];
  const float pz = xyz1[(b * 3 + 2) * N_ + n];

  // top-3 smallest squared distances (sorted ascending)
  float d0 = 3.4e38f, d1 = 3.4e38f, d2 = 3.4e38f;
  int   i0 = 0, i1 = 0, i2 = 0;
  for (int s = 0; s < S_; ++s) {
    float dx = px - sx[s], dy = py - sy[s], dz = pz - sz[s];
    float d  = dx * dx + dy * dy + dz * dz;
    if (d < d0)      { d2 = d1; i2 = i1; d1 = d0; i1 = i0; d0 = d; i0 = s; }
    else if (d < d1) { d2 = d1; i2 = i1; d1 = d;  i1 = s; }
    else if (d < d2) { d2 = d;  i2 = s; }
  }
  float w0 = 1.0f / (d0 + 1e-8f);
  float w1 = 1.0f / (d1 + 1e-8f);
  float w2 = 1.0f / (d2 + 1e-8f);
  const float wsum = w0 + w1 + w2;
  w0 /= wsum; w1 /= wsum; w2 /= wsum;
  idx3[p * 3 + 0] = i0; idx3[p * 3 + 1] = i1; idx3[p * 3 + 2] = i2;
  w3[p * 3 + 0] = w0;  w3[p * 3 + 1] = w1;  w3[p * 3 + 2] = w2;

  // nearest keypoint (coords are elements 1..3 of each feature row)
  float dm = 3.4e38f; int im = 0;
  for (int m = 0; m < M_; ++m) {
    float dx = px - sf[m * 4 + 1], dy = py - sf[m * 4 + 2], dz = pz - sf[m * 4 + 3];
    float d  = dx * dx + dy * dy + dz * dz;
    if (d < dm) { dm = d; im = m; }
  }
  __bf16* xr = Xbf + (size_t)p * KPAD0;
  xr[384] = (__bf16)sf[im * 4 + 0];
  xr[385] = (__bf16)sf[im * 4 + 1];
  xr[386] = (__bf16)sf[im * 4 + 2];
  xr[387] = (__bf16)sf[im * 4 + 3];
  for (int j = CIN_; j < KPAD0; ++j) xr[j] = (__bf16)0.0f;
}

// ---------------------------------------------------------------------------
// Kernel 2: build X channels 0..383 = [points1 (128) | 3-NN interp (256)].
// One block per point, one thread per channel.
// ---------------------------------------------------------------------------
__global__ void concat_kernel(const float* __restrict__ points1,
                              const float* __restrict__ points2,
                              const int* __restrict__ idx3,
                              const float* __restrict__ w3,
                              __bf16* __restrict__ Xbf) {
  const int p = blockIdx.x;
  const int b = p >> 12;       // N_ = 4096
  const int n = p & (N_ - 1);
  const int c = threadIdx.x;   // 0..383
  float v;
  if (c < D1_) {
    v = points1[((size_t)b * D1_ + c) * N_ + n];
  } else {
    const int c2 = c - D1_;
    const float* pr = points2 + ((size_t)b * D2_ + c2) * S_;
    v = w3[p * 3 + 0] * pr[idx3[p * 3 + 0]] +
        w3[p * 3 + 1] * pr[idx3[p * 3 + 1]] +
        w3[p * 3 + 2] * pr[idx3[p * 3 + 2]];
  }
  Xbf[(size_t)p * KPAD0 + c] = (__bf16)v;
}

// ---------------------------------------------------------------------------
// Weight convert f32 [Cout, Cin] -> bf16 [Cout, Kp] zero-padded.
// ---------------------------------------------------------------------------
__global__ void wconv_kernel(const float* __restrict__ W, __bf16* __restrict__ Wb,
                             int Cout, int Cin, int Kp) {
  const int g = blockIdx.x * 256 + threadIdx.x;
  if (g >= Cout * Kp) return;
  const int c = g / Kp, k = g % Kp;
  Wb[g] = (k < Cin) ? (__bf16)W[(size_t)c * Cin + k] : (__bf16)0.0f;
}

// ---------------------------------------------------------------------------
// WMMA GEMM with LDS-staged activations.
//   Y[p, m] = sum_k W[m, k] * X[p, k]
// Block: MGROUPS waves; all waves share one 32-point x K activation tile in
// LDS (row stride K+8 bf16 -> lane LDS banks hit 4*i mod 64, conflict-free).
// Wave w handles output channels [w*64, w*64+64) for both 16-point halves:
// 4 m-tiles x 2 p-tiles = 8 accumulators; A frag feeds 2 WMMAs, B frag 4.
// Fragment layouts per CDNA5 ISA 7.12.2:
//   A 16x32 bf16: kb = 8*(lane>>4); elems 0..7 = K kb..kb+7,
//                 elems 8..15 = K kb+16..kb+23   (two 16B global loads)
//   B 32x16 bf16: kb = 8*(lane>>4); elems 0..7 = K kb..kb+7,
//                 elems 8..15 = K kb+16..kb+23   (two 16B LDS loads)
//   D f32 16x16 : VGPR r -> m = 8*(lane>>4)+r, n = lane&15 (one 32B store)
// ---------------------------------------------------------------------------
__device__ __forceinline__ v16bf combine16(v8bf lo, v8bf hi) {
  v16bf f;
#pragma unroll
  for (int i = 0; i < 8; ++i) { f[i] = lo[i]; f[i + 8] = hi[i]; }
  return f;
}

template <int K, int MGROUPS>
__global__ __launch_bounds__(MGROUPS * 32)
void gemm_wmma_lds_kernel(const __bf16* __restrict__ X,
                          const __bf16* __restrict__ W,
                          float* __restrict__ Y) {
  constexpr int LDSK = K + 8;          // pad 16B: conflict-free ds_load_b128
  constexpr int COUT = MGROUPS * 64;
  __shared__ __bf16 sX[32 * LDSK];

  const int tid  = threadIdx.x;
  const int lane = tid & 31;
  const int mg   = tid >> 5;           // wave id == m-group (64 channels)
  const int p0   = blockIdx.x * 32;    // 32 points per block

  // cooperative stage of X tile [32 x K] into LDS
  for (int ch = tid; ch < 32 * (K / 8); ch += MGROUPS * 32) {
    const int row = ch / (K / 8);
    const int col = (ch - row * (K / 8)) * 8;
    *(v8bf*)(sX + row * LDSK + col) =
        *(const v8bf*)(X + (size_t)(p0 + row) * K + col);
  }
  __syncthreads();

  const int half = lane >> 4;
  const int l16  = lane & 15;
  const __bf16* s0 = sX + (size_t)l16 * LDSK + (half << 3);         // p-tile 0
  const __bf16* s1 = sX + (size_t)(16 + l16) * LDSK + (half << 3);  // p-tile 1
  const __bf16* wbase = W + (size_t)((mg << 6) + l16) * K + (half << 3);

  const v8f z = {0.f, 0.f, 0.f, 0.f, 0.f, 0.f, 0.f, 0.f};
  v8f acc[4][2];
#pragma unroll
  for (int mt = 0; mt < 4; ++mt) { acc[mt][0] = z; acc[mt][1] = z; }

#pragma unroll 4
  for (int kk = 0; kk < K; kk += 32) {
    const v16bf b0 = combine16(*(const v8bf*)(s0 + kk), *(const v8bf*)(s0 + kk + 16));
    const v16bf b1 = combine16(*(const v8bf*)(s1 + kk), *(const v8bf*)(s1 + kk + 16));
#pragma unroll
    for (int mt = 0; mt < 4; ++mt) {
      const __bf16* wr = wbase + (size_t)(mt << 4) * K + kk;
      const v16bf af = combine16(*(const v8bf*)(wr), *(const v8bf*)(wr + 16));
      acc[mt][0] = __builtin_amdgcn_wmma_f32_16x16x32_bf16(
          false, af, false, b0, (short)0, acc[mt][0], false, false);
      acc[mt][1] = __builtin_amdgcn_wmma_f32_16x16x32_bf16(
          false, af, false, b1, (short)0, acc[mt][1], false, false);
    }
  }

#pragma unroll
  for (int mt = 0; mt < 4; ++mt) {
#pragma unroll
    for (int ps = 0; ps < 2; ++ps) {
      const int p = p0 + (ps << 4) + l16;
      float* yb = Y + (size_t)p * COUT + (mg << 6) + (mt << 4) + (half << 3);
      *(v8f*)yb = acc[mt][ps];
    }
  }
}

// ---------------------------------------------------------------------------
// BN reduce: per-channel mean and rsqrt(var+eps) over all P samples.
// Block = 16 channels x 16 p-lanes (channel reads coalesced).
// ---------------------------------------------------------------------------
__global__ void bn_reduce_kernel(const float* __restrict__ Y, int Cout,
                                 float* __restrict__ meanArr,
                                 float* __restrict__ invArr) {
  __shared__ float ssum[256], ssq[256];
  const int tid  = threadIdx.x;
  const int cidx = tid & 15;
  const int pl   = tid >> 4;
  const int c    = blockIdx.x * 16 + cidx;
  float s = 0.f, q = 0.f;
  for (int p = pl; p < P_; p += 16) {
    float v = Y[(size_t)p * Cout + c];
    s += v; q += v * v;
  }
  ssum[tid] = s; ssq[tid] = q;
  __syncthreads();
  for (int off = 128; off >= 16; off >>= 1) {
    if (tid < off) { ssum[tid] += ssum[tid + off]; ssq[tid] += ssq[tid + off]; }
    __syncthreads();
  }
  if (tid < 16) {
    const float mean = ssum[tid] * (1.0f / (float)P_);
    const float var  = ssq[tid] * (1.0f / (float)P_) - mean * mean;
    meanArr[c] = mean;
    invArr[c]  = rsqrtf(var + BN_EPS_);
  }
}

// BN apply + ReLU -> bf16 activations [P, Cout] for the next layer.
__global__ void bn_apply_kernel(const float* __restrict__ Y, int Cout,
                                const float* __restrict__ g,
                                const float* __restrict__ be,
                                const float* __restrict__ meanArr,
                                const float* __restrict__ invArr,
                                __bf16* __restrict__ Out) {
  const size_t gid = (size_t)blockIdx.x * 256 + threadIdx.x;
  const size_t i4  = gid * 4;
  const int c = (int)(i4 % (size_t)Cout);
  const float4 y = *(const float4*)(Y + i4);
  v4bf o;
  o[0] = (__bf16)fmaxf((y.x - meanArr[c + 0]) * invArr[c + 0] * g[c + 0] + be[c + 0], 0.f);
  o[1] = (__bf16)fmaxf((y.y - meanArr[c + 1]) * invArr[c + 1] * g[c + 1] + be[c + 1], 0.f);
  o[2] = (__bf16)fmaxf((y.z - meanArr[c + 2]) * invArr[c + 2] * g[c + 2] + be[c + 2], 0.f);
  o[3] = (__bf16)fmaxf((y.w - meanArr[c + 3]) * invArr[c + 3] * g[c + 3] + be[c + 3], 0.f);
  *(v4bf*)(Out + i4) = o;
}

// Final BN apply + ReLU -> f32 output in reference layout [B, 256, N].
__global__ void bn_final_kernel(const float* __restrict__ Y,
                                const float* __restrict__ g,
                                const float* __restrict__ be,
                                const float* __restrict__ meanArr,
                                const float* __restrict__ invArr,
                                float* __restrict__ Out) {
  const int o = blockIdx.x * 256 + threadIdx.x;
  const int b = o >> 20;            // 256 * 4096 = 1<<20
  const int c = (o >> 12) & 255;
  const int n = o & (N_ - 1);
  const int p = (b << 12) + n;
  const float v = (Y[(size_t)p * C2_ + c] - meanArr[c]) * invArr[c] * g[c] + be[c];
  Out[o] = fmaxf(v, 0.f);
}

// ---------------------------------------------------------------------------
extern "C" void kernel_launch(void* const* d_in, const int* in_sizes, int n_in,
                              void* d_out, int out_size, void* d_ws, size_t ws_size,
                              hipStream_t stream) {
  (void)in_sizes; (void)n_in; (void)out_size; (void)ws_size;
  const float* xyz1    = (const float*)d_in[0];
  const float* xyz2    = (const float*)d_in[1];
  const float* points1 = (const float*)d_in[2];
  const float* points2 = (const float*)d_in[3];
  const float* feature = (const float*)d_in[4];
  const float* W0 = (const float*)d_in[5];
  const float* g0 = (const float*)d_in[7];
  const float* be0 = (const float*)d_in[8];
  const float* W1 = (const float*)d_in[9];
  const float* g1 = (const float*)d_in[11];
  const float* be1 = (const float*)d_in[12];
  const float* W2 = (const float*)d_in[13];
  const float* g2 = (const float*)d_in[15];
  const float* be2 = (const float*)d_in[16];

  char* ws = (char*)d_ws;
  size_t off = 0;
  auto carve = [&](size_t bytes) -> void* {
    void* pp = ws + off;
    off = (off + bytes + 255) & ~(size_t)255;
    return pp;
  };
  __bf16* Xbf  = (__bf16*)carve((size_t)P_ * KPAD0 * 2);
  __bf16* act1 = (__bf16*)carve((size_t)P_ * C0_ * 2);
  __bf16* act2 = (__bf16*)carve((size_t)P_ * C1_ * 2);
  float*  yf   = (float*) carve((size_t)P_ * C0_ * 4);
  __bf16* W0b  = (__bf16*)carve((size_t)C0_ * KPAD0 * 2);
  __bf16* W1b  = (__bf16*)carve((size_t)C1_ * C0_ * 2);
  __bf16* W2b  = (__bf16*)carve((size_t)C2_ * C1_ * 2);
  int*    idx3 = (int*)   carve((size_t)P_ * 3 * 4);
  float*  w3   = (float*) carve((size_t)P_ * 3 * 4);
  float*  mean = (float*) carve(512 * 4);
  float*  inv  = (float*) carve(512 * 4);

  // weights -> bf16 (zero padded K)
  wconv_kernel<<<(C0_ * KPAD0 + 255) / 256, 256, 0, stream>>>(W0, W0b, C0_, CIN_, KPAD0);
  wconv_kernel<<<(C1_ * C0_   + 255) / 256, 256, 0, stream>>>(W1, W1b, C1_, C0_, C0_);
  wconv_kernel<<<(C2_ * C1_   + 255) / 256, 256, 0, stream>>>(W2, W2b, C2_, C1_, C1_);

  // build X
  knn_prep_kernel<<<B_ * (N_ / 256), 256, 0, stream>>>(xyz1, xyz2, feature, Xbf, idx3, w3);
  concat_kernel<<<P_, 384, 0, stream>>>(points1, points2, idx3, w3, Xbf);

  // layer 0: 512 x 416 GEMM + BN + ReLU
  gemm_wmma_lds_kernel<KPAD0, 8><<<P_ / 32, 256, 0, stream>>>(Xbf, W0b, yf);
  bn_reduce_kernel<<<C0_ / 16, 256, 0, stream>>>(yf, C0_, mean, inv);
  bn_apply_kernel<<<(size_t)P_ * C0_ / 1024, 256, 0, stream>>>(yf, C0_, g0, be0, mean, inv, act1);

  // layer 1: 512 x 512
  gemm_wmma_lds_kernel<C0_, 8><<<P_ / 32, 256, 0, stream>>>(act1, W1b, yf);
  bn_reduce_kernel<<<C1_ / 16, 256, 0, stream>>>(yf, C1_, mean, inv);
  bn_apply_kernel<<<(size_t)P_ * C1_ / 1024, 256, 0, stream>>>(yf, C1_, g1, be1, mean, inv, act2);

  // layer 2: 256 x 512 -> output [B, 256, N] f32
  gemm_wmma_lds_kernel<C1_, 4><<<P_ / 32, 128, 0, stream>>>(act2, W2b, yf);
  bn_reduce_kernel<<<C2_ / 16, 256, 0, stream>>>(yf, C2_, mean, inv);
  bn_final_kernel<<<(size_t)P_ * C2_ / 256, 256, 0, stream>>>(yf, g2, be2, mean, inv, (float*)d_out);
}
